// GroupAttention_5317169512522
// MI455X (gfx1250) — matrix operand order repaired and checked by reading, not compiled, not run on hardware
//
#include <hip/hip_runtime.h>

// ---------------------------------------------------------------------------
// Fused windowed attention for MI455X (gfx1250, wave32, WMMA).
// Per 64-token window: x -> bf16 LDS -> QKV WMMA -> QK^T WMMA -> softmax
// (shfl+exp) -> P*V WMMA -> proj WMMA + bias -> f32 out.
// All matmuls: v_wmma_f32_16x16x32_bf16 (f32 accumulate).
// f32x2 -> bf16x2 via round-add + v_perm_b32 (3 VALU per 2 elements).
// ---------------------------------------------------------------------------

typedef __attribute__((ext_vector_type(16))) __bf16 v16bf;
typedef __attribute__((ext_vector_type(2)))  __bf16 v2bf;
typedef __attribute__((ext_vector_type(8)))  float  v8f;

#define CC   128   // channels
#define NH   4     // heads
#define HD   32    // head dim
#define S64  64    // tokens per window
#define LDX  136   // LDS stride for X / attention-out (64 x 128, padded)
#define LDQK 264   // LDS stride for Q|K buffer (64 x 256, padded)
#define LDVT 72    // LDS stride for V^T buffer (128 x 64, padded)
#define LDP  72    // LDS stride for P buffer (per head 64 x 64, padded)

#define OFF_X   0                         // 64*136*2  = 17408  (aliased as sO)
#define OFF_QK  17408                     // 64*264*2  = 33792
#define OFF_VT  (17408 + 33792)           // 128*72*2  = 18432
#define OFF_P   (17408 + 33792 + 18432)   // 4*64*72*2 = 36864
#define SMEM_BYTES (17408 + 33792 + 18432 + 36864)  // 106496 B < 320 KB/WGP

// packed f32x2 -> bf16x2: round-to-nearest (ties up) then one v_perm_b32
// merging the two high halves.  result[0] = cvt(a), result[1] = cvt(b).
__device__ __forceinline__ v2bf pk2(float a, float b) {
  unsigned ua = __builtin_bit_cast(unsigned, a) + 0x8000u;
  unsigned ub = __builtin_bit_cast(unsigned, b) + 0x8000u;
  // src = (ub << 32) | ua ; dst bytes = {ub.b3, ub.b2, ua.b3, ua.b2}
  unsigned r = __builtin_amdgcn_perm(ub, ua, 0x07060302u);
  return __builtin_bit_cast(v2bf, r);
}

// A fragment (16x32 bf16, M=lane&15): 16 contiguous halves per lane.
// p = base + (m0 + (lane&15))*ld + k0 + ((lane>>4)&1)*16
__device__ __forceinline__ v16bf load_frag_a(const __bf16* p) {
  v16bf a;
#pragma unroll
  for (int e = 0; e < 16; ++e) a[e] = p[e];
  return a;
}

// B fragment (32x16 bf16, N=lane&15) from a K-contiguous (B^T row-major)
// buffer: 16 contiguous halves per lane.
// p = base + (n0 + (lane&15))*ld + k0 + ((lane>>4)&1)*16
__device__ __forceinline__ v16bf load_frag_bt(const __bf16* p) {
  v16bf b;
#pragma unroll
  for (int e = 0; e < 16; ++e) b[e] = p[e];
  return b;
}

__device__ __forceinline__ v8f wmma_bf16(v16bf a, v16bf b, v8f c) {
  return __builtin_amdgcn_wmma_f32_16x16x32_bf16(false, a, false, b,
                                                 (short)0, c, false, false);
}

// ---------------------------------------------------------------------------
// Prologue: convert weights f32 -> bf16 once (then L2-resident).
// ---------------------------------------------------------------------------
__global__ __launch_bounds__(256) void cvt_weights(const float* __restrict__ wqkv,
                                                   const float* __restrict__ wproj,
                                                   __bf16* __restrict__ wq_bf,
                                                   __bf16* __restrict__ wp_bf) {
  const int i = (blockIdx.x * 256 + threadIdx.x) * 2;   // pairs
  if (i < 3 * CC * CC) {
    *(v2bf*)(wq_bf + i) = pk2(wqkv[i], wqkv[i + 1]);
  }
  if (i < CC * CC) {
    *(v2bf*)(wp_bf + i) = pk2(wproj[i], wproj[i + 1]);
  }
}

// ---------------------------------------------------------------------------
// Main fused kernel: one block per (batch, window), 8 waves.
// ---------------------------------------------------------------------------
__global__ __launch_bounds__(256) void win_attn(const float* __restrict__ x,
                                                const __bf16* __restrict__ wqkv,
                                                const __bf16* __restrict__ wproj,
                                                const float* __restrict__ bproj,
                                                float* __restrict__ out) {
  __shared__ __align__(16) unsigned char smem[SMEM_BYTES];
  __bf16* sX  = (__bf16*)(smem + OFF_X);   // 64 x LDX   (window of x, bf16)
  __bf16* sQK = (__bf16*)(smem + OFF_QK);  // 64 x LDQK  (cols 0..127 Q, 128..255 K)
  __bf16* sVt = (__bf16*)(smem + OFF_VT);  // 128 x LDVT (V^T: [head*32+d][t])
  __bf16* sP  = (__bf16*)(smem + OFF_P);   // NH x 64 x LDP (softmax probs)
  __bf16* sO  = sX;                        // attention output aliases sX

  const int bg   = blockIdx.x;
  const int b    = bg >> 6;
  const int g    = bg & 63;
  const int hi   = g >> 3, wi = g & 7;     // window position in 8x8 grid
  const int tid  = threadIdx.x;
  const int lane = tid & 31;
  const int wave = tid >> 5;
  const int nlo  = lane & 15;
  const int khalf = (lane >> 4) & 1;

  // ---- Phase 1: stage window of x into LDS as bf16 ----------------------
  for (int i = tid; i < S64 * 32; i += 256) {        // 64 rows * 32 float4
    const int s  = i >> 5;
    const int c4 = (i & 31) << 2;
    const int n  = (hi * 8 + (s >> 3)) * 64 + wi * 8 + (s & 7);
    const float4 v = *(const float4*)(x + ((size_t)(b * 4096 + n)) * CC + c4);
    __bf16* dst = sX + s * LDX + c4;
    *(v2bf*)(dst + 0) = pk2(v.x, v.y);
    *(v2bf*)(dst + 2) = pk2(v.z, v.w);
  }
  __syncthreads();

  // ---- Phase 2: QKV GEMM  qkv[s,o] = sum_c X[s,c] * w_qkv[o,c] ----------
  // wave -> fixed row-tile (wave&3), 12 col-tiles; A fragments loaded once.
  {
    const int m0    = (wave & 3) * 16;
    const int obase = (wave >> 2) * 12;
    v16bf afrag[4];
#pragma unroll
    for (int kt = 0; kt < 4; ++kt)
      afrag[kt] = load_frag_a(sX + (m0 + nlo) * LDX + kt * 32 + khalf * 16);
    for (int ot = obase; ot < obase + 12; ++ot) {
      const int o0 = ot * 16;
      v8f acc = {0, 0, 0, 0, 0, 0, 0, 0};
#pragma unroll
      for (int kt = 0; kt < 4; ++kt)
        acc = wmma_bf16(afrag[kt],
                        load_frag_bt(wqkv + (size_t)(o0 + nlo) * CC + kt * 32 + khalf * 16),
                        acc);
      if (o0 < 2 * CC) {              // Q and K: row-major [token][chan]
#pragma unroll
        for (int r = 0; r < 8; r += 2) {
          v2bf p = pk2(acc[r], acc[r + 1]);
          sQK[(m0 + r     + khalf * 8) * LDQK + o0 + nlo] = p[0];
          sQK[(m0 + r + 1 + khalf * 8) * LDQK + o0 + nlo] = p[1];
        }
      } else {                        // V: transposed [chan][token]
        const int dd = o0 - 2 * CC + nlo;
        __bf16* vp = sVt + dd * LDVT + m0 + khalf * 8;   // even offset -> 4B aligned
#pragma unroll
        for (int r = 0; r < 8; r += 2)
          *(v2bf*)(vp + r) = pk2(acc[r], acc[r + 1]);
      }
    }
  }
  __syncthreads();

  // ---- Phase 3: per-head attention. wave -> (head = wave/2, row half) ----
  {
    const int h     = wave >> 1;
    const int rbase = (wave & 1) * 32;          // rows rbase..rbase+31
    const float scale = 0.17677669529663687f;   // 32^-0.5
    __bf16* myP = sP + h * (S64 * LDP);

    // S = Q K^T : 2 row-tiles x 4 col-tiles, K=32 -> 1 WMMA per tile.
    v16bf qfrag[2];
#pragma unroll
    for (int mt = 0; mt < 2; ++mt)
      qfrag[mt] = load_frag_a(sQK + (rbase + mt * 16 + nlo) * LDQK + h * HD + khalf * 16);
    v8f sacc[2][4];
#pragma unroll
    for (int nt = 0; nt < 4; ++nt) {
      v16bf kb = load_frag_bt(sQK + (nt * 16 + nlo) * LDQK + CC + h * HD + khalf * 16);
#pragma unroll
      for (int mt = 0; mt < 2; ++mt) {
        v8f z = {0, 0, 0, 0, 0, 0, 0, 0};
        sacc[mt][nt] = wmma_bf16(qfrag[mt], kb, z);
      }
    }

    // Row softmax: reduce across nt and the 16 lanes sharing a row.
#pragma unroll
    for (int mt = 0; mt < 2; ++mt) {
#pragma unroll
      for (int r = 0; r < 8; ++r) {
        float v0 = sacc[mt][0][r] * scale;
        float v1 = sacc[mt][1][r] * scale;
        float v2 = sacc[mt][2][r] * scale;
        float v3 = sacc[mt][3][r] * scale;
        float mx = fmaxf(fmaxf(v0, v1), fmaxf(v2, v3));
#pragma unroll
        for (int m = 1; m <= 8; m <<= 1) mx = fmaxf(mx, __shfl_xor(mx, m, 32));
        float e0 = __expf(v0 - mx), e1 = __expf(v1 - mx);
        float e2 = __expf(v2 - mx), e3 = __expf(v3 - mx);
        float sm = (e0 + e1) + (e2 + e3);
#pragma unroll
        for (int m = 1; m <= 8; m <<= 1) sm += __shfl_xor(sm, m, 32);
        const float inv = 1.0f / sm;
        const int row = rbase + mt * 16 + r + khalf * 8;
        __bf16* pp = myP + row * LDP;
        v2bf p01 = pk2(e0 * inv, e1 * inv);
        v2bf p23 = pk2(e2 * inv, e3 * inv);
        pp[ 0 + nlo] = p01[0];
        pp[16 + nlo] = p01[1];
        pp[32 + nlo] = p23[0];
        pp[48 + nlo] = p23[1];
      }
    }
    // (no barrier: wave reads back only its own sP rows; DS is in-order)

    // O_head = P @ V : 2x2 tiles of 16x16, K=64 -> 2 WMMA per tile.
    v16bf pfrag[2][2];   // [mt][kt]
#pragma unroll
    for (int mt = 0; mt < 2; ++mt)
#pragma unroll
      for (int kt = 0; kt < 2; ++kt)
        pfrag[mt][kt] =
            load_frag_a(myP + (rbase + mt * 16 + nlo) * LDP + kt * 32 + khalf * 16);
    v16bf vfrag[2][2];   // [nt][kt]
#pragma unroll
    for (int nt = 0; nt < 2; ++nt)
#pragma unroll
      for (int kt = 0; kt < 2; ++kt)
        vfrag[nt][kt] =
            load_frag_bt(sVt + (h * HD + nt * 16 + nlo) * LDVT + kt * 32 + khalf * 16);
#pragma unroll
    for (int mt = 0; mt < 2; ++mt)
#pragma unroll
      for (int nt = 0; nt < 2; ++nt) {
        v8f acc = {0, 0, 0, 0, 0, 0, 0, 0};
        acc = wmma_bf16(pfrag[mt][0], vfrag[nt][0], acc);
        acc = wmma_bf16(pfrag[mt][1], vfrag[nt][1], acc);
        const int col = h * HD + nt * 16 + nlo;
#pragma unroll
        for (int r = 0; r < 8; r += 2) {   // sX dead after phase 2 -> safe alias
          v2bf p = pk2(acc[r], acc[r + 1]);
          sO[(rbase + mt * 16 + r     + khalf * 8) * LDX + col] = p[0];
          sO[(rbase + mt * 16 + r + 1 + khalf * 8) * LDX + col] = p[1];
        }
      }
  }
  __syncthreads();

  // ---- Phase 4: fused projection  y[s,o] = sum_c O[s,c]*w_proj[o,c] + b --
  // wave -> fixed row-tile (wave&3), 4 col-tiles; A fragments loaded once.
  {
    const int m0    = (wave & 3) * 16;
    const int obase = (wave >> 2) * 4;
    v16bf afrag[4];
#pragma unroll
    for (int kt = 0; kt < 4; ++kt)
      afrag[kt] = load_frag_a(sO + (m0 + nlo) * LDX + kt * 32 + khalf * 16);
#pragma unroll
    for (int ot = obase; ot < obase + 4; ++ot) {
      const int o0 = ot * 16;
      v8f acc = {0, 0, 0, 0, 0, 0, 0, 0};
#pragma unroll
      for (int kt = 0; kt < 4; ++kt)
        acc = wmma_bf16(afrag[kt],
                        load_frag_bt(wproj + (size_t)(o0 + nlo) * CC + kt * 32 + khalf * 16),
                        acc);
      const float bias = bproj[o0 + nlo];
#pragma unroll
      for (int r = 0; r < 8; ++r) {
        const int s = m0 + r + khalf * 8;
        const int n = (hi * 8 + (s >> 3)) * 64 + wi * 8 + (s & 7);   // un-window
        out[((size_t)(b * 4096 + n)) * CC + o0 + nlo] = acc[r] + bias;
      }
    }
  }
}

// ---------------------------------------------------------------------------
extern "C" void kernel_launch(void* const* d_in, const int* in_sizes, int n_in,
                              void* d_out, int out_size, void* d_ws, size_t ws_size,
                              hipStream_t stream) {
  (void)in_sizes; (void)n_in; (void)out_size; (void)ws_size;
  const float* x     = (const float*)d_in[0];   // [32,4096,128] f32
  const float* wqkv  = (const float*)d_in[1];   // [384,128] f32
  const float* wproj = (const float*)d_in[2];   // [128,128] f32
  const float* bproj = (const float*)d_in[3];   // [128] f32
  // d_in[4], d_in[5] are H=64, W=64 scalars (fixed, unused)
  float* out = (float*)d_out;

  __bf16* wq_bf = (__bf16*)d_ws;                // 3*128*128 bf16
  __bf16* wp_bf = wq_bf + 3 * CC * CC;          // 128*128 bf16

  cvt_weights<<<dim3((3 * CC * CC / 2 + 255) / 256), dim3(256), 0, stream>>>(
      wqkv, wproj, wq_bf, wp_bf);

  win_attn<<<dim3(32 * 64), dim3(256), 0, stream>>>(x, wq_bf, wp_bf, bproj, out);
}
